// NN_54657753809054
// MI455X (gfx1250) — matrix operand order repaired and checked by reading, not compiled, not run on hardware
//
#include <hip/hip_runtime.h>
#include <stdint.h>

typedef __attribute__((ext_vector_type(16))) _Float16 v16h;
typedef __attribute__((ext_vector_type(8)))  _Float16 v8h;
typedef __attribute__((ext_vector_type(8)))  float    v8f;
typedef __attribute__((ext_vector_type(4)))  uint32_t u32x4;
typedef __attribute__((ext_vector_type(8)))  int32_t  i32x8;
typedef __attribute__((ext_vector_type(4)))  int32_t  i32x4;

#define B_SZ   1024
#define IN_SZ  256
#define HID_SZ 512
#define BTILE  16

// ---------- counter-based RNG: PCG hash + Box-Muller ----------
__device__ __forceinline__ uint32_t pcg_hash(uint32_t v) {
    v = v * 747796405u + 2891336453u;
    v = ((v >> ((v >> 28) + 4u)) ^ v) * 277803737u;
    return (v >> 22) ^ v;
}
__device__ __forceinline__ float gauss_at(uint32_t idx, uint32_t salt) {
    uint32_t h1 = pcg_hash(idx ^ salt);
    uint32_t h2 = pcg_hash(h1 + 0x9E3779B9u);
    float u1 = (float)(h1 >> 8) * 0x1p-24f + 0x1p-25f;   // (0,1)
    float u2 = (float)(h2 >> 8) * 0x1p-24f;              // [0,1)
    return sqrtf(-2.0f * __logf(u1)) * __cosf(6.283185307179586f * u2);
}

// ---------- WMMA helper (codegen-confirmed builtin) ----------
__device__ __forceinline__ v8f wmma_acc(v16h a, v16h b, v8f c) {
    return __builtin_amdgcn_wmma_f32_16x16x32_f16(
        /*neg_a=*/false, a, /*neg_b=*/false, b,
        /*c_mod=*/(short)0, c, /*reuse_a=*/false, /*reuse_b=*/false);
}

// A-operand gather: two contiguous 8-half runs (16B each) -> v16h
__device__ __forceinline__ v16h make_a(const _Float16* p0, const _Float16* p1) {
    v8h a = *(const v8h*)p0;
    v8h b = *(const v8h*)p1;
    v16h r;
#pragma unroll
    for (int i = 0; i < 8; ++i) { r[i] = a[i]; r[i + 8] = b[i]; }
    return r;
}

// ---------- prep: f32 -> (hi,lo) f16 splits; exp(logvar) ----------
__global__ __launch_bounds__(256) void bnn_prep(
    const float* __restrict__ x, const float* __restrict__ w1mu,
    const float* __restrict__ w1lv,
    _Float16* __restrict__ Xpack,     // [Xhi | Xlo | X2hi | X2lo], each B*IN
    _Float16* __restrict__ WMhi, _Float16* __restrict__ WMlo,
    _Float16* __restrict__ WVhi, _Float16* __restrict__ WVlo)
{
    const int i = blockIdx.x * 256 + threadIdx.x;
    if (i < B_SZ * IN_SZ) {
        float v = x[i];
        _Float16 hi = (_Float16)v;
        Xpack[i]                     = hi;
        Xpack[B_SZ * IN_SZ + i]      = (_Float16)(v - (float)hi);
        float s = v * v;
        _Float16 sh = (_Float16)s;
        Xpack[2 * B_SZ * IN_SZ + i]  = sh;
        Xpack[3 * B_SZ * IN_SZ + i]  = (_Float16)(s - (float)sh);
    }
    if (i < HID_SZ * IN_SZ) {
        float m = w1mu[i];
        _Float16 mh = (_Float16)m;
        WMhi[i] = mh;
        WMlo[i] = (_Float16)(m - (float)mh);
        float vv = __expf(w1lv[i]);
        _Float16 vh = (_Float16)vv;
        WVhi[i] = vh;
        WVlo[i] = (_Float16)(vv - (float)vh);
    }
}

// ---------- fused BNN kernel: one 16-row batch tile per workgroup ----------
__global__ __launch_bounds__(256) void bnn_fused(
    const _Float16* __restrict__ Xpack,
    const _Float16* __restrict__ WMhi, const _Float16* __restrict__ WMlo,
    const _Float16* __restrict__ WVhi, const _Float16* __restrict__ WVlo,
    const float* __restrict__ b1mu, const float* __restrict__ b1lv,
    const float* __restrict__ w2mu, const float* __restrict__ w2lv,
    const float* __restrict__ b2mu, const float* __restrict__ b2lv,
    float* __restrict__ out)
{
    __shared__ __align__(16) _Float16 ldsA[4][BTILE][IN_SZ]; // 32 KB x-tile splits
    __shared__ float ldsH[BTILE][HID_SZ];                    // 32 KB hidden acts

    const int tid  = threadIdx.x;
    const int wave = tid >> 5;
    const int lane = tid & 31;
    const int b0   = blockIdx.x * BTILE;

    // Stage x hi/lo and x^2 hi/lo tiles into LDS.
#if __has_builtin(__builtin_amdgcn_tensor_load_to_lds) && __has_builtin(__builtin_amdgcn_s_wait_tensorcnt)
    // Tensor Data Mover: one 2D descriptor moves the whole 32KB tile.
    // Tensor: 4 rows (the 4 split arrays) x 4096 f16 elements, row stride = B*IN.
    if (tid < 32) {   // one wave issues the DMA (TDM ignores EXEC; avoid 8x issue)
        const uint64_t gaddr = (uint64_t)(uintptr_t)Xpack +
                               (uint64_t)b0 * (IN_SZ * 2);          // tile start
        const uint32_t lds_off = (uint32_t)(uintptr_t)(&ldsA[0][0][0]);
        const u32x4 g0 = {
            1u,                                                     // count=1 (user D#)
            lds_off,                                                // lds_addr
            (uint32_t)(gaddr & 0xFFFFFFFFu),                        // global_addr[31:0]
            (uint32_t)((gaddr >> 32) & 0x1FFFFFFu) | (2u << 30)     // ga[56:32] | type=2
        };
        const i32x8 g1 = {
            (int)(1u << 16),              // data_size=1 (2 bytes), no pad/iter/multicast
            (int)((BTILE * IN_SZ) << 16), // tensor_dim0 = 4096 (low 16 bits)
            (int)(4u << 16),              // tensor_dim0[31:16]=0 | tensor_dim1 = 4
            (int)((BTILE * IN_SZ) << 16), // tensor_dim1[31:16]=0 | tile_dim0 = 4096
            4,                            // tile_dim1 = 4, tile_dim2 = 0
            (int)(B_SZ * IN_SZ),          // tensor_dim0_stride = 262144 elements
            0, 0                          // stride hi / tensor_dim1_stride (unused, 2D)
        };
        const i32x4 gz4 = {0, 0, 0, 0};
        const i32x8 gz8 = {0, 0, 0, 0, 0, 0, 0, 0};
        __builtin_amdgcn_tensor_load_to_lds(g0, g1, gz4, gz4, gz8, 0);
        __builtin_amdgcn_s_wait_tensorcnt(0);                       // TENSORcnt == 0
    }
#else
    {
        const uint4* src = (const uint4*)Xpack;   // 16B = 8 halves
        uint4* dst = (uint4*)(&ldsA[0][0][0]);
        for (int i = tid; i < 4 * 512; i += 256) {
            const int arr = i >> 9;
            const int off = i & 511;
            dst[i] = src[arr * (B_SZ * IN_SZ / 8) + b0 * (IN_SZ / 8) + off];
        }
    }
#endif
    __syncthreads();

    v8f accM[4] = {};   // mean GEMM accumulators (4 N-tiles per wave)
    v8f accV[4] = {};   // variance GEMM accumulators

    const int m   = lane & 15;          // A row / B column within tile
    const int kb  = (lane >> 4) * 8;    // unused (A and B both index with kb2)
    const int kb2 = (lane >> 4) * 16;   // A and B run base: k = 16*(lane/16) + s

    for (int kk = 0; kk < IN_SZ; kk += 32) {
        v16h aMh = make_a(&ldsA[0][m][kk + kb2], &ldsA[0][m][kk + kb2 + 8]);
        v16h aMl = make_a(&ldsA[1][m][kk + kb2], &ldsA[1][m][kk + kb2 + 8]);
        v16h aVh = make_a(&ldsA[2][m][kk + kb2], &ldsA[2][m][kk + kb2 + 8]);
        v16h aVl = make_a(&ldsA[3][m][kk + kb2], &ldsA[3][m][kk + kb2 + 8]);
#pragma unroll
        for (int t = 0; t < 4; ++t) {
            const int n    = (wave * 4 + t) * 16 + m;  // hidden unit (B column)
            const int woff = n * IN_SZ + kk + kb2;
            v16h bMh = *(const v16h*)(WMhi + woff);
            v16h bMl = *(const v16h*)(WMlo + woff);
            v16h bVh = *(const v16h*)(WVhi + woff);
            v16h bVl = *(const v16h*)(WVlo + woff);
            // Markidis split: hi*hi + hi*lo + lo*hi (~f32 accuracy)
            accM[t] = wmma_acc(aMh, bMh, accM[t]);
            accM[t] = wmma_acc(aMh, bMl, accM[t]);
            accM[t] = wmma_acc(aMl, bMh, accM[t]);
            accV[t] = wmma_acc(aVh, bVh, accV[t]);
            accV[t] = wmma_acc(aVh, bVl, accV[t]);
            accV[t] = wmma_acc(aVl, bVh, accV[t]);
        }
    }

    // Epilogue: bias fold, local-reparam noise, ReLU, stage h to LDS.
#pragma unroll
    for (int t = 0; t < 4; ++t) {
        const int n  = (wave * 4 + t) * 16 + m;
        const float bm = b1mu[n];
        const float bv = __expf(b1lv[n]);
#pragma unroll
        for (int r = 0; r < 8; ++r) {
            const int mrow = r + ((lane >> 4) << 3);   // C/D layout: M = r + 8*(lane/16)
            const int b    = b0 + mrow;
            float mean1 = accM[t][r] + bm;
            float var1  = fmaxf(accV[t][r] + bv, 0.0f);
            float h = mean1 + sqrtf(var1) * gauss_at((uint32_t)(b * HID_SZ + n), 0xB5297A4Du);
            ldsH[mrow][n] = fmaxf(h, 0.0f);
        }
    }
    __syncthreads();

    // Layer 2 (OUT=2): per batch row, 4 length-512 dot products via lane-strided
    // accumulation + wave shuffle reduction. Each wave owns 2 rows.
#pragma unroll
    for (int rr = 0; rr < 2; ++rr) {
        const int row = wave * 2 + rr;
        const int b   = b0 + row;
        float am0 = 0.f, am1 = 0.f, av0 = 0.f, av1 = 0.f;
        for (int j = lane; j < HID_SZ; j += 32) {
            const float h  = ldsH[row][j];
            const float h2 = h * h;
            am0 += h  * w2mu[j];
            am1 += h  * w2mu[HID_SZ + j];
            av0 += h2 * __expf(w2lv[j]);
            av1 += h2 * __expf(w2lv[HID_SZ + j]);
        }
#pragma unroll
        for (int off = 16; off > 0; off >>= 1) {
            am0 += __shfl_xor(am0, off, 32);
            am1 += __shfl_xor(am1, off, 32);
            av0 += __shfl_xor(av0, off, 32);
            av1 += __shfl_xor(av1, off, 32);
        }
        if (lane == 0) {
            float o0 = am0 + b2mu[0];
            float v0 = fmaxf(av0 + __expf(b2lv[0]), 0.0f);
            o0 += sqrtf(v0) * gauss_at((uint32_t)(b * 2 + 0), 0x68E31DA4u);
            float o1 = am1 + b2mu[1];
            float v1 = fmaxf(av1 + __expf(b2lv[1]), 0.0f);
            o1 += sqrtf(v1) * gauss_at((uint32_t)(b * 2 + 1), 0x68E31DA4u);
            out[b]        = o0;                              // mean
            out[B_SZ + b] = fmaxf(__expf(o1), 1e-6f);        // var
        }
    }
}

extern "C" void kernel_launch(void* const* d_in, const int* in_sizes, int n_in,
                              void* d_out, int out_size, void* d_ws, size_t ws_size,
                              hipStream_t stream) {
    const float* x    = (const float*)d_in[0];
    const float* w1mu = (const float*)d_in[1];
    const float* w1lv = (const float*)d_in[2];
    const float* b1mu = (const float*)d_in[3];
    const float* b1lv = (const float*)d_in[4];
    const float* w2mu = (const float*)d_in[5];
    const float* w2lv = (const float*)d_in[6];
    const float* b2mu = (const float*)d_in[7];
    const float* b2lv = (const float*)d_in[8];
    float* out = (float*)d_out;

    char* ws = (char*)d_ws;
    _Float16* Xpack = (_Float16*)ws;                              // 4 * 512KB = 2MB
    _Float16* WMhi  = (_Float16*)(ws + 4u * B_SZ * IN_SZ * 2u);   // 4 * 256KB = 1MB
    _Float16* WMlo  = WMhi + HID_SZ * IN_SZ;
    _Float16* WVhi  = WMlo + HID_SZ * IN_SZ;
    _Float16* WVlo  = WVhi + HID_SZ * IN_SZ;

    bnn_prep<<<(B_SZ * IN_SZ + 255) / 256, 256, 0, stream>>>(
        x, w1mu, w1lv, Xpack, WMhi, WMlo, WVhi, WVlo);
    bnn_fused<<<B_SZ / BTILE, 256, 0, stream>>>(
        Xpack, WMhi, WMlo, WVhi, WVlo,
        b1mu, b1lv, w2mu, w2lv, b2mu, b2lv, out);
}
